// LSTMCell_75797582840479
// MI455X (gfx1250) — matrix operand order repaired and checked by reading, not compiled, not run on hardware
//
#include <hip/hip_runtime.h>
#include <math.h>

typedef __attribute__((ext_vector_type(16))) __bf16 v16bf;
typedef __attribute__((ext_vector_type(8)))  __bf16 v8bf;
typedef __attribute__((ext_vector_type(8)))  float  v8f;

#define B_DIM 4096
#define D_DIM 2048
#define U_DIM 2048
#define K_DIM 4096   // D + U
#define N_DIM 8192   // 4 * U

#define BM 128       // rows per workgroup
#define BU 64        // u columns per workgroup (x4 gates -> 256 n')
#define BN 256       // total n' per workgroup
#define KT 64        // K slice per stage
#define NK (K_DIM / KT)
#define LDSS 72      // padded LDS stride (bf16 elements)
#define ABUF_B (BM * LDSS * 2)          // 18432 B : A tile 128 x 64
#define WBUF_B (BN * LDSS * 2)          // 36864 B : W tile 256(n') x 64(k)
#define STAGE_B (ABUF_B + WBUF_B)       // 55296 B per buffer
#define SMEM_B (BM * BN * 4)            // 131072 B (z spill; >= 2*STAGE_B)

__device__ __forceinline__ unsigned short f2bf(float x) {
    unsigned int b = __float_as_uint(x);
    b += 0x7FFFu + ((b >> 16) & 1u);          // round-to-nearest-even
    return (unsigned short)(b >> 16);
}

// ---- pack [inputs | h_tm1] -> Abf16 [4096 x 4096] row-major ----
__global__ __launch_bounds__(256) void cvtA(const float* __restrict__ inp,
                                            const float* __restrict__ h,
                                            unsigned short* __restrict__ Abf) {
    long long flat = ((long long)blockIdx.x * 256 + threadIdx.x) * 4;
    int b = (int)(flat >> 12);
    int k = (int)(flat & 4095);
    const float* src = (k < D_DIM) ? (inp + (size_t)b * D_DIM + k)
                                   : (h   + (size_t)b * U_DIM + (k - D_DIM));
    float4 v = *(const float4*)src;
    uint2 o;
    o.x = (unsigned)f2bf(v.x) | ((unsigned)f2bf(v.y) << 16);
    o.y = (unsigned)f2bf(v.z) | ((unsigned)f2bf(v.w) << 16);
    *(uint2*)(Abf + flat) = o;
}

// ---- pack+transpose [kernel ; recurrent_kernel] -> WbfT [8192(n) x 4096(k)] ----
__global__ __launch_bounds__(256) void cvtWT(const float* __restrict__ kern,
                                             const float* __restrict__ rec,
                                             unsigned short* __restrict__ WbfT) {
    __shared__ unsigned short tile[32][33];
    const int k0 = blockIdx.x * 32;
    const int n0 = blockIdx.y * 32;
    const int t  = threadIdx.x;
#pragma unroll
    for (int i = 0; i < 4; ++i) {
        int id = i * 256 + t;
        int kl = id >> 5, nl = id & 31;     // read coalesced along n
        int k = k0 + kl;
        float v = (k < D_DIM) ? kern[(size_t)k * N_DIM + n0 + nl]
                              : rec[(size_t)(k - D_DIM) * N_DIM + n0 + nl];
        tile[kl][nl] = f2bf(v);
    }
    __syncthreads();
#pragma unroll
    for (int i = 0; i < 4; ++i) {
        int id = i * 256 + t;
        int nl = id >> 5, kl = id & 31;     // write coalesced along k
        WbfT[(size_t)(n0 + nl) * K_DIM + k0 + kl] = tile[kl][nl];
    }
}

__device__ __forceinline__ float hsig(float x) {
    return fminf(fmaxf(0.2f * x + 0.5f, 0.0f), 1.0f);
}

// ---- fused bf16-WMMA GEMM + LSTM epilogue (double-buffered LDS, 64x64 wave tiles) ----
__global__ __launch_bounds__(256)
void lstm_gemm(const unsigned short* __restrict__ Abf,
               const unsigned short* __restrict__ WbfT,
               const float* __restrict__ c_tm1,
               float* __restrict__ out) {
    extern __shared__ char smem[];
    float* zbuf = (float*)smem;             // 128 x 256 f32 (reuses staging LDS)

    const int tid  = threadIdx.x;
    const int lane = tid & 31;
    const int wave = tid >> 5;
    const int wm0  = (wave & 1) * 64;       // wave row offset in tile (2 groups)
    const int wn0  = (wave >> 1) * 64;      // wave n' offset in tile (4 groups)
    const int m0   = blockIdx.x * BM;
    const int u0   = blockIdx.y * BU;

    const int rsel = lane & 15;
    const int kbA  = (lane < 16) ? 0 : 8;   // A frag k-base (ISA 7.12.2, 16-bit A 16x32)
    const int kbB  = (lane < 16) ? 0 : 8;   // B frag k-base (same k order as A)

    // staging chunk decode: A = 128 rows x 8 chunks, W = 256 n'-rows x 8 chunks
    size_t gAoff[4], gWoff[8];
    int    ldsAoff[4], ldsWoff[8];
#pragma unroll
    for (int it = 0; it < 4; ++it) {
        int chunk = it * 256 + tid;
        int row = chunk >> 3;
        int c8  = (chunk & 7) * 8;
        gAoff[it]   = (size_t)(m0 + row) * K_DIM + c8;
        ldsAoff[it] = row * LDSS + c8;
    }
#pragma unroll
    for (int it = 0; it < 8; ++it) {
        int chunk = it * 256 + tid;
        int row = chunk >> 3;                       // n' 0..255
        int c8  = (chunk & 7) * 8;
        int ng  = (row >> 6) * U_DIM + u0 + (row & 63);   // n' -> global n
        gWoff[it]   = (size_t)ng * K_DIM + c8;
        ldsWoff[it] = row * LDSS + c8;
    }

    v8f acc[4][4];
    const v8f vzero = {0.f, 0.f, 0.f, 0.f, 0.f, 0.f, 0.f, 0.f};
#pragma unroll
    for (int mi = 0; mi < 4; ++mi)
#pragma unroll
        for (int ni = 0; ni < 4; ++ni) acc[mi][ni] = vzero;

    uint4 ra[4], rw[8];
    // prime buffer 0 with K-slice 0
#pragma unroll
    for (int it = 0; it < 4; ++it) ra[it] = *(const uint4*)(Abf  + gAoff[it]);
#pragma unroll
    for (int it = 0; it < 8; ++it) rw[it] = *(const uint4*)(WbfT + gWoff[it]);
    {
        unsigned short* dA = (unsigned short*)(smem);
        unsigned short* dW = (unsigned short*)(smem + ABUF_B);
#pragma unroll
        for (int it = 0; it < 4; ++it) *(uint4*)(dA + ldsAoff[it]) = ra[it];
#pragma unroll
        for (int it = 0; it < 8; ++it) *(uint4*)(dW + ldsWoff[it]) = rw[it];
    }
    __syncthreads();

    for (int kt = 0; kt < NK; ++kt) {
        const int cur = kt & 1;
        const int nxt = cur ^ 1;
        const bool more = (kt + 1) < NK;

        // issue next tile's global loads first (overlap with WMMA below)
        if (more) {
            const size_t koff = (size_t)(kt + 1) * KT;
#pragma unroll
            for (int it = 0; it < 4; ++it)
                ra[it] = *(const uint4*)(Abf  + gAoff[it] + koff);
#pragma unroll
            for (int it = 0; it < 8; ++it)
                rw[it] = *(const uint4*)(WbfT + gWoff[it] + koff);
        }

        const unsigned short* smA = (const unsigned short*)(smem + cur * STAGE_B);
        const unsigned short* smW = (const unsigned short*)(smem + cur * STAGE_B + ABUF_B);

#pragma unroll
        for (int ks = 0; ks < KT; ks += 32) {
            v16bf a[4], b[4];
#pragma unroll
            for (int mi = 0; mi < 4; ++mi) {
                const unsigned short* p = smA + (wm0 + mi * 16 + rsel) * LDSS + ks + kbA;
                union { v16bf v; v8bf h[2]; } u;
                u.h[0] = *(const v8bf*)p;          // k = kbA .. kbA+7
                u.h[1] = *(const v8bf*)(p + 16);   // k = kbA+16 .. kbA+23
                a[mi] = u.v;
            }
#pragma unroll
            for (int ni = 0; ni < 4; ++ni) {
                const unsigned short* p = smW + (wn0 + ni * 16 + rsel) * LDSS + ks + kbB;
                union { v16bf v; v8bf h[2]; } u;
                u.h[0] = *(const v8bf*)p;          // k = kbB .. kbB+7
                u.h[1] = *(const v8bf*)(p + 16);   // k = kbB+16 .. kbB+23
                b[ni] = u.v;
            }
#pragma unroll
            for (int mi = 0; mi < 4; ++mi)
#pragma unroll
                for (int ni = 0; ni < 4; ++ni)
                    acc[mi][ni] = __builtin_amdgcn_wmma_f32_16x16x32_bf16(
                        false, a[mi], false, b[ni], (short)0, acc[mi][ni], false, false);
        }

        // store next tile into the other buffer (readers of it finished last iter)
        if (more) {
            unsigned short* dA = (unsigned short*)(smem + nxt * STAGE_B);
            unsigned short* dW = (unsigned short*)(smem + nxt * STAGE_B + ABUF_B);
#pragma unroll
            for (int it = 0; it < 4; ++it) *(uint4*)(dA + ldsAoff[it]) = ra[it];
#pragma unroll
            for (int it = 0; it < 8; ++it) *(uint4*)(dW + ldsWoff[it]) = rw[it];
        }
        __syncthreads();
    }

    // spill z fragments to LDS (C/D layout: lane<16 -> M=r, lane>=16 -> M=8+r)
    const int mo = (lane >> 4) << 3;
#pragma unroll
    for (int mi = 0; mi < 4; ++mi)
#pragma unroll
        for (int ni = 0; ni < 4; ++ni) {
            int n = wn0 + ni * 16 + rsel;
#pragma unroll
            for (int r = 0; r < 8; ++r)
                zbuf[(wm0 + mi * 16 + mo + r) * BN + n] = acc[mi][ni][r];
        }
    __syncthreads();

    // fused LSTM elementwise epilogue: 128 rows x 64 u-cells
    float* out_h1 = out;
    float* out_h2 = out + (size_t)B_DIM * U_DIM;
    float* out_c  = out + (size_t)2 * B_DIM * U_DIM;
#pragma unroll
    for (int it = 0; it < 32; ++it) {
        int cell = it * 256 + tid;
        int m = cell >> 6;
        int j = cell & 63;
        float zi = zbuf[m * BN +       j];
        float zf = zbuf[m * BN +  64 + j];
        float zc = zbuf[m * BN + 128 + j];
        float zo = zbuf[m * BN + 192 + j];
        size_t o = (size_t)(m0 + m) * U_DIM + (u0 + j);
        float cp = c_tm1[o];
        float cv = hsig(zf) * cp + hsig(zi) * tanhf(zc);
        float hv = hsig(zo) * tanhf(cv);
        out_h1[o] = hv;
        out_h2[o] = hv;
        out_c[o]  = cv;
    }
}

extern "C" void kernel_launch(void* const* d_in, const int* in_sizes, int n_in,
                              void* d_out, int out_size, void* d_ws, size_t ws_size,
                              hipStream_t stream) {
    const float* inputs = (const float*)d_in[0];
    const float* h_tm1  = (const float*)d_in[1];
    const float* c_tm1  = (const float*)d_in[2];
    const float* kern   = (const float*)d_in[3];
    const float* rec    = (const float*)d_in[4];

    unsigned short* Abf  = (unsigned short*)d_ws;                             // 32 MiB
    unsigned short* WbfT = (unsigned short*)((char*)d_ws +
                                             (size_t)B_DIM * K_DIM * 2);      // 64 MiB

    // allow 128 KB dynamic LDS (no-op if already permitted)
    (void)hipFuncSetAttribute((const void*)lstm_gemm,
                              hipFuncAttributeMaxDynamicSharedMemorySize, SMEM_B);

    cvtA<<<(B_DIM * K_DIM) / (256 * 4), 256, 0, stream>>>(inputs, h_tm1, Abf);
    dim3 tgrid(K_DIM / 32, N_DIM / 32);  // 128 x 256
    cvtWT<<<tgrid, 256, 0, stream>>>(kern, rec, WbfT);

    dim3 grid(B_DIM / BM, U_DIM / BU);   // 32 x 32
    lstm_gemm<<<grid, 256, SMEM_B, stream>>>(Abf, WbfT, c_tm1, (float*)d_out);
}
